// SphConv_31190052503574
// MI455X (gfx1250) — compile-verified, matrix operand order checked by reading, not run on hardware
//
#include <hip/hip_runtime.h>
#include <hip/hip_bf16.h>
#include <math.h>

// ---------------------------------------------------------------------------
// SphConv (DimeNet-style) for MI455X / gfx1250.
//   E = num edges (in_sizes[2]), T = num triplets (in_sizes[4])
//   out = [ h (E*128) | rbf (E*6) | sbf (T*42) ]  (float32, return order)
// Roofline: sbf writes ~T*168B ~ 390MB dominate (17us @ 23.3TB/s) -> NT stores.
// GEMM E x 384 x 128 (~9.7 GFLOP) -> f16 WMMA (v_wmma_f32_16x16x32_f16).
// A chunks staged into LDS with the Tensor Data Mover (double buffered),
// B pre-packed into WMMA fragment layout -> pure b128 loads.
// ---------------------------------------------------------------------------

typedef __attribute__((ext_vector_type(16))) _Float16 v16h;
typedef __attribute__((ext_vector_type(8)))  _Float16 v8h;
typedef __attribute__((ext_vector_type(8)))  float    v8f;
typedef __attribute__((ext_vector_type(4)))  unsigned int u32x4;
typedef __attribute__((ext_vector_type(8)))  int      i32x8;
typedef __attribute__((ext_vector_type(4)))  int      i32x4;

union V16U { v16h v; v8h h[2]; };

#define HIDDEN   128
#define NSPH     7
#define NRAD     6
#define RBFS_PAD 48      // 42 used, padded to 48 floats (192B) for b128 loads
#define GROWS    32      // A rows per chunk (32*384*2B = 24KB); 2 chunks in LDS

#ifndef USE_TDM
#if __has_builtin(__builtin_amdgcn_tensor_load_to_lds)
#define USE_TDM 1
#else
#define USE_TDM 0
#endif
#endif

// ---------------- spherical Bessel (device) ----------------

__device__ __forceinline__ double jl_d(int l, double x) {
  double s = sin(x), c = cos(x);
  double j0 = s / x;
  if (l == 0) return j0;
  double jm = j0, jc = s / (x * x) - c / x;
  for (int ll = 1; ll < l; ++ll) {
    double jn = (2.0 * ll + 1.0) / x * jc - jm;
    jm = jc; jc = jn;
  }
  return jc;
}

// Roots z_{l,n} of j_l (interlacing + bisection, as in reference) and
// NORM = sqrt(2)/|j_{l+1}(z)|.  One thread; negligible; deterministic.
__global__ void bessel_const_kernel(float* __restrict__ Zb, float* __restrict__ Nb) {
  if (threadIdx.x != 0 || blockIdx.x != 0) return;
  double prev[NSPH + NRAD], cur[NSPH + NRAD];
  int len = NRAD + NSPH - 1; // 12 zeros of j0 to start
  for (int t = 0; t < len; ++t) prev[t] = M_PI * (double)(t + 1);
  for (int r = 0; r < NRAD; ++r) Zb[r] = (float)prev[r];
  for (int l = 1; l < NSPH; ++l) {
    for (int t = 0; t < len - 1; ++t) {
      double lo = prev[t], hi = prev[t + 1];
      double flo = jl_d(l, lo);
      for (int it = 0; it < 80; ++it) {
        double mid = 0.5 * (lo + hi);
        double fm = jl_d(l, mid);
        if ((fm >= 0.0) == (flo >= 0.0)) { lo = mid; flo = fm; }
        else { hi = mid; }
      }
      cur[t] = 0.5 * (lo + hi);
    }
    len--;
    for (int t = 0; t < len; ++t) prev[t] = cur[t];
    for (int r = 0; r < NRAD; ++r) Zb[l * NRAD + r] = (float)prev[r];
  }
  for (int l = 0; l < NSPH; ++l)
    for (int r = 0; r < NRAD; ++r) {
      double jn = jl_d(l + 1, (double)Zb[l * NRAD + r]);
      Nb[l * NRAD + r] = (float)(sqrt(2.0) / fabs(jn));
    }
}

// ---------------- per-edge radial / spherical basis ----------------

__global__ __launch_bounds__(256)
void edge_kernel(const float* __restrict__ pos, const int* __restrict__ vi,
                 const int* __restrict__ vj, const float* __restrict__ freq,
                 const float* __restrict__ Zb, const float* __restrict__ Nb,
                 float* __restrict__ rbf, float* __restrict__ rbfs, int E) {
  int e = blockIdx.x * blockDim.x + threadIdx.x;
  if (e >= E) return;
  int a = vi[e], b = vj[e];
  float dx = pos[3 * a] - pos[3 * b];
  float dy = pos[3 * a + 1] - pos[3 * b + 1];
  float dz = pos[3 * a + 2] - pos[3 * b + 2];
  float d = sqrtf(dx * dx + dy * dy + dz * dz) * 0.2f; // /CUTOFF
  // DimeNet envelope, P = 6: 1/x - 28 x^5 + 48 x^6 - 21 x^7, zero for x >= 1
  float x2 = d * d;
  float x5 = x2 * x2 * d;
  float env = (d < 1.0f)
                  ? (1.0f / d + x5 * (-28.0f + d * (48.0f - 21.0f * d)))
                  : 0.0f;
#pragma unroll
  for (int r = 0; r < NRAD; ++r)
    rbf[(size_t)e * NRAD + r] = env * sinf(freq[r] * d);

  float* out = rbfs + (size_t)e * RBFS_PAD;
#pragma unroll
  for (int l = 0; l < NSPH; ++l) {
#pragma unroll
    for (int r = 0; r < NRAD; ++r) {
      float t = d * Zb[l * NRAD + r];
      float s = sinf(t), c = cosf(t);
      float jc = s / t;
      if (l > 0) {
        float jm = jc;
        jc = s / (t * t) - c / t;
        for (int ll = 1; ll < l; ++ll) {
          float jn = (float)(2 * ll + 1) / t * jc - jm;
          jm = jc; jc = jn;
        }
      }
      out[l * NRAD + r] = env * Nb[l * NRAD + r] * jc;
    }
  }
}

// ---------------- pack A = [x_i | x_j | swish(rbf @ Wr + br)] as f16 ----------------

__global__ __launch_bounds__(128)
void abuild_kernel(const int* __restrict__ z, const int* __restrict__ vi,
                   const int* __restrict__ vj, const float* __restrict__ emb_w,
                   const float* __restrict__ wr, const float* __restrict__ br,
                   const float* __restrict__ rbf, _Float16* __restrict__ Af, int E) {
  int e = blockIdx.x;
  if (e >= E) return;
  int t = threadIdx.x;
  int zi = z[vi[e]], zj = z[vj[e]];
  _Float16* row = Af + (size_t)e * 384;
  row[t] = (_Float16)emb_w[(size_t)zi * HIDDEN + t];
  row[HIDDEN + t] = (_Float16)emb_w[(size_t)zj * HIDDEN + t];
  float s = br[t];
#pragma unroll
  for (int r = 0; r < NRAD; ++r)
    s += rbf[(size_t)e * NRAD + r] * wr[r * HIDDEN + t];
  s = s * (1.0f / (1.0f + __expf(-s)));
  row[256 + t] = (_Float16)s;
}

// ---------------- pack W[384,128] into WMMA B-fragment layout ----------------
// Wp[((ct*12 + kk)*32 + lane)*16 + t] so each lane's B fragment is one
// contiguous 32B chunk -> two global_load_b128, zero half-shuffles.

__global__ __launch_bounds__(256)
void wpack_kernel(const float* __restrict__ w, _Float16* __restrict__ wp) {
  int idx = blockIdx.x * blockDim.x + threadIdx.x; // 8*12*32 = 3072
  if (idx >= 8 * 12 * 32) return;
  int lane = idx & 31;
  int kk = (idx >> 5) % 12;
  int ct = idx / (12 * 32);
  int lr = lane & 15, hi = lane >> 4;
  int col = ct * 16 + lr;
  _Float16* dst = wp + (size_t)idx * 16;
#pragma unroll
  for (int t = 0; t < 8; ++t) {
    dst[t]     = (_Float16)w[(kk * 32 + hi * 8 + t) * HIDDEN + col];
    dst[8 + t] = (_Float16)w[(kk * 32 + 16 + hi * 8 + t) * HIDDEN + col];
  }
}

// ---------------- TDM issue helper ----------------
// D# per CDNA5 ISA: group0 = {count=1, lds_addr,
// global_addr, type=2}; group1 = {data_size=2B, tensor_dim0/1, tile_dim0/1,
// tensor_dim0_stride}.  1-row tile of tile_elems halves -> plain linear DMA.
// This toolchain declares the 6-arg builtin:
//   (u32x4 g0, i32x8 g1, i32x4 g2, i32x4 g3, i32x8 extra, i32 cpol)

#if USE_TDM
__device__ __forceinline__ void tdm_load_chunk(const void* gptr, unsigned lds_off,
                                               unsigned tile_elems,
                                               unsigned tensor_elems) {
  unsigned long long ga = (unsigned long long)gptr;
  u32x4 g0;
  g0[0] = 1u;                                   // count=1, user mode
  g0[1] = lds_off;                              // lds_addr (bytes)
  g0[2] = (unsigned)(ga & 0xffffffffu);         // global_addr[31:0]
  g0[3] = (unsigned)((ga >> 32) & 0x01ffffffu)  // global_addr[56:32]
          | (2u << 30);                         // type = 2 (image)
  i32x8 g1;
  g1[0] = (int)(1u << 16);                              // data_size = 2 bytes
  g1[1] = (int)((tensor_elems & 0xffffu) << 16);        // tensor_dim0[15:0]
  g1[2] = (int)((tensor_elems >> 16) | (1u << 16));     // tensor_dim0[31:16] | tensor_dim1=1
  g1[3] = (int)((tile_elems & 0xffffu) << 16);          // tile_dim0
  g1[4] = 1;                                            // tile_dim1 = 1
  g1[5] = (int)tensor_elems;                            // tensor_dim0_stride lo
  g1[6] = 0;
  g1[7] = 0;
  i32x4 gz4 = {0, 0, 0, 0};
  i32x8 gz8 = {0, 0, 0, 0, 0, 0, 0, 0};
  __builtin_amdgcn_tensor_load_to_lds(g0, g1, gz4, gz4, gz8, 0);
}
#endif

// ---------------- WMMA GEMM: h = swish(A[E,384] @ W[384,128] + bias) ----------------
// block = 256 threads = 8 wave32s; wave w owns column tile [16w, 16w+16).
// Two GROWS-row chunks per block, TDM double-buffered into LDS.

__global__ __launch_bounds__(256)
void gemm_kernel(const _Float16* __restrict__ Af, const _Float16* __restrict__ Wp,
                 const float* __restrict__ bias, float* __restrict__ h, int E) {
  extern __shared__ _Float16 As[]; // 2 * GROWS*384 halves
  const int row0 = blockIdx.x * (2 * GROWS);
  const int wave = threadIdx.x >> 5;
  const int lane = threadIdx.x & 31;
  const int lr = lane & 15;
  const int hi = lane >> 4;
  const int col = wave * 16 + lr;

  // B fragments: contiguous pre-packed 32B chunks per (kk, lane).
  V16U b[12];
  const _Float16* wp = Wp + ((size_t)(wave * 12) * 32 + lane) * 16;
#pragma unroll
  for (int kk = 0; kk < 12; ++kk) {
    b[kk].h[0] = *(const v8h*)(wp + kk * 32 * 16);
    b[kk].h[1] = *(const v8h*)(wp + kk * 32 * 16 + 8);
  }
  const float bv = bias[col];

#if USE_TDM
  if (wave == 0) {
    tdm_load_chunk(Af + (size_t)row0 * 384, 0u,
                   GROWS * 384, (unsigned)E * 384u);
    tdm_load_chunk(Af + (size_t)(row0 + GROWS) * 384,
                   (unsigned)(GROWS * 384 * sizeof(_Float16)),
                   GROWS * 384, (unsigned)E * 384u);
    __builtin_amdgcn_s_wait_tensorcnt(1); // chunk 0 landed (in-order)
  }
  __syncthreads();
#else
  {
    const uint4* src = (const uint4*)(Af + (size_t)row0 * 384);
    uint4* dst = (uint4*)As;
    const size_t lim = ((size_t)E * 384) >> 3;
    const size_t base = ((size_t)row0 * 384) >> 3;
    for (int idx = threadIdx.x; idx < (2 * GROWS * 384) / 8; idx += 256)
      if (base + idx < lim) dst[idx] = src[idx];
  }
  __syncthreads();
#endif

  for (int c = 0; c < 2; ++c) {
    const _Float16* chunk = As + c * (GROWS * 384);
#pragma unroll
    for (int rt = 0; rt < GROWS / 16; ++rt) {
      v8f acc = {};
      const _Float16* ap = chunk + (rt * 16 + lr) * 384;
#pragma unroll
      for (int kk = 0; kk < 12; ++kk) {
        V16U a;
        // 16-bit A 16x32 layout: lanes 0-15 halves0-7 = K0..7 (K16..23 hi),
        // lanes 16-31 shifted by 8 in K.  Two ds_load_b128 per fragment.
        a.h[0] = *(const v8h*)(ap + kk * 32 + hi * 8);
        a.h[1] = *(const v8h*)(ap + kk * 32 + 16 + hi * 8);
        acc = __builtin_amdgcn_wmma_f32_16x16x32_f16(
            /*neg_a=*/false, a.v, /*neg_b=*/false, b[kk].v,
            /*c_mod=*/(short)0, acc, /*reuse_a=*/false, /*reuse_b=*/false);
      }
      // D layout: VGPR v -> row v (lanes 0-15) / row v+8 (lanes 16-31).
      const int gr0 = row0 + c * GROWS + rt * 16 + hi * 8;
#pragma unroll
      for (int v = 0; v < 8; ++v) {
        int gr = gr0 + v;
        if (gr < E) {
          float val = acc[v] + bv;
          val = val * (1.0f / (1.0f + __expf(-val)));
          h[(size_t)gr * HIDDEN + col] = val;
        }
      }
    }
#if USE_TDM
    if (c == 0) {
      if (wave == 0) __builtin_amdgcn_s_wait_tensorcnt(0); // chunk 1 landed
      __syncthreads();
    }
#endif
  }
}

// ---------------- triplet sbf: bandwidth-bound, NT stores ----------------

__global__ __launch_bounds__(256)
void sbf_kernel(const float* __restrict__ pos, const int* __restrict__ ti,
                const int* __restrict__ tj, const int* __restrict__ tk,
                const int* __restrict__ tkj, const float* __restrict__ rbfs,
                float* __restrict__ sbf, int T) {
  int t = blockIdx.x * blockDim.x + threadIdx.x;
  if (t >= T) return;
  int ii = ti[t], jj = tj[t], kk = tk[t], e = tkj[t];
  float ix = pos[3 * ii], iy = pos[3 * ii + 1], iz = pos[3 * ii + 2];
  float jx = pos[3 * jj] - ix, jy = pos[3 * jj + 1] - iy, jz = pos[3 * jj + 2] - iz;
  float kx = pos[3 * kk] - ix, ky = pos[3 * kk + 1] - iy, kz = pos[3 * kk + 2] - iz;
  float a = jx * kx + jy * ky + jz * kz;
  float cx = jy * kz - jz * ky;
  float cy = jz * kx - jx * kz;
  float cz = jx * ky - jy * kx;
  float bb = sqrtf(cx * cx + cy * cy + cz * cz);
  float ct = cosf(atan2f(bb, a)); // matches reference semantics (incl. 0,0)

  float P[NSPH];
  P[0] = 1.0f;
  P[1] = ct;
#pragma unroll
  for (int l = 1; l < NSPH - 1; ++l)
    P[l + 1] = ((float)(2 * l + 1) * ct * P[l] - (float)l * P[l - 1]) / (float)(l + 1);

  const float YC[NSPH] = {0.28209479177387814f, 0.4886025119029199f,
                          0.6307831305050401f,  0.7463526651802308f,
                          0.8462843753216345f,  0.9356025796273889f,
                          1.0171072362820548f};
  const float* row = rbfs + (size_t)e * RBFS_PAD;
  float* out = sbf + (size_t)t * (NSPH * NRAD);
#pragma unroll
  for (int l = 0; l < NSPH; ++l) {
    float cb = P[l] * YC[l];
#pragma unroll
    for (int r = 0; r < NRAD; ++r)
      __builtin_nontemporal_store(row[l * NRAD + r] * cb, out + l * NRAD + r);
  }
}

// ---------------- host launcher ----------------

extern "C" void kernel_launch(void* const* d_in, const int* in_sizes, int n_in,
                              void* d_out, int out_size, void* d_ws, size_t ws_size,
                              hipStream_t stream) {
  const int*   z         = (const int*)d_in[0];
  const float* pos       = (const float*)d_in[1];
  const int*   vi        = (const int*)d_in[2];
  const int*   vj        = (const int*)d_in[3];
  const int*   ti        = (const int*)d_in[4];
  const int*   tj        = (const int*)d_in[5];
  const int*   tk        = (const int*)d_in[6];
  const int*   tkj       = (const int*)d_in[7];
  // d_in[8] = idx_ji (unused by the outputs)
  const float* freq      = (const float*)d_in[9];
  const float* emb_w     = (const float*)d_in[10];
  const float* lin_rbf_w = (const float*)d_in[11];
  const float* lin_rbf_b = (const float*)d_in[12];
  const float* lin_w     = (const float*)d_in[13];
  const float* lin_b     = (const float*)d_in[14];

  const int E = in_sizes[2]; // edges
  const int T = in_sizes[4]; // triplets

  float* h_out   = (float*)d_out;
  float* rbf_out = h_out + (size_t)E * HIDDEN;
  float* sbf_out = rbf_out + (size_t)E * NRAD;

  // workspace layout
  float* wsf  = (float*)d_ws;
  float* Zb   = wsf;        // 64 floats (42 used)
  float* Nb   = wsf + 64;   // 64 floats (42 used)
  float* rbfs = wsf + 128;  // E * RBFS_PAD floats
  _Float16* Af = (_Float16*)(rbfs + (size_t)E * RBFS_PAD); // E*384 halves
  _Float16* Wp = Af + (size_t)E * 384;                     // 8*12*32*16 halves

  bessel_const_kernel<<<1, 1, 0, stream>>>(Zb, Nb);

  edge_kernel<<<(E + 255) / 256, 256, 0, stream>>>(pos, vi, vj, freq, Zb, Nb,
                                                   rbf_out, rbfs, E);

  wpack_kernel<<<(8 * 12 * 32 + 255) / 256, 256, 0, stream>>>(lin_w, Wp);

  abuild_kernel<<<E, 128, 0, stream>>>(z, vi, vj, emb_w, lin_rbf_w, lin_rbf_b,
                                       rbf_out, Af, E);

  const int gblocks = (E + 2 * GROWS - 1) / (2 * GROWS);
  gemm_kernel<<<gblocks, 256, 2 * GROWS * 384 * sizeof(_Float16), stream>>>(
      Af, Wp, lin_b, h_out, E);

  sbf_kernel<<<(T + 255) / 256, 256, 0, stream>>>(pos, ti, tj, tk, tkj, rbfs,
                                                  sbf_out, T);
}
